// MultiHeadAttention_38053410242897
// MI455X (gfx1250) — matrix operand order repaired and checked by reading, not compiled, not run on hardware
//
#include <hip/hip_runtime.h>

// Problem constants (reference: B=2, S=4096, D=512, H=8, DH=64)
#define BATCH   2
#define S_LEN   4096
#define D_MODEL 512
#define N_HEADS 8
#define D_HEAD  64

typedef __attribute__((ext_vector_type(16))) _Float16 v16h;
typedef __attribute__((ext_vector_type(8)))  _Float16 v8h;
typedef __attribute__((ext_vector_type(8)))  float    v8f;
typedef __attribute__((ext_vector_type(4)))  float    v4f;

__device__ __forceinline__ v8f wmma_f16(v16h a, v16h b, v8f c) {
  // D = A(16x32 f16) * B(32x16 f16) + C(16x16 f32)
  return __builtin_amdgcn_wmma_f32_16x16x32_f16(false, a, false, b, (short)0, c, false, false);
}

// B-fragment: lane L = column n (L%16), elems 0..7 <-> K = 8h + e, elems 8..15 <-> K = 16 + 8h + e.
// p = (row-base + 8*h); two 8-f16 loads at p and p + 16.
__device__ __forceinline__ v16h load_b16(const _Float16* __restrict__ p) {
  v8h lo = *(const v8h*)(p);
  v8h hi = *(const v8h*)(p + 16);
  v16h r;
#pragma unroll
  for (int e = 0; e < 8; ++e) { r[e] = lo[e]; r[8 + e] = hi[e]; }
  return r;
}

// A-fragment (f16 source): lane L = row m (L%16); elems 0..7 <-> K=kb+8h+e,
// elems 8..15 <-> K=kb+16+8h+e  (h = L/16). Two 16B contiguous loads.
__device__ __forceinline__ v16h load_a16_h(const _Float16* __restrict__ row, int kb, int h) {
  v8h lo = *(const v8h*)(row + kb + 8 * h);
  v8h hi = *(const v8h*)(row + kb + 16 + 8 * h);
  v16h r;
#pragma unroll
  for (int e = 0; e < 8; ++e) { r[e] = lo[e]; r[8 + e] = hi[e]; }
  return r;
}

// A-fragment from f32 source, convert to f16 on the fly.
__device__ __forceinline__ v16h load_a16_f(const float* __restrict__ row, int kb, int h) {
  v4f a0 = *(const v4f*)(row + kb + 8 * h);
  v4f a1 = *(const v4f*)(row + kb + 8 * h + 4);
  v4f b0 = *(const v4f*)(row + kb + 16 + 8 * h);
  v4f b1 = *(const v4f*)(row + kb + 16 + 8 * h + 4);
  v16h r;
#pragma unroll
  for (int e = 0; e < 4; ++e) {
    r[e]      = (_Float16)a0[e];
    r[4 + e]  = (_Float16)a1[e];
    r[8 + e]  = (_Float16)b0[e];
    r[12 + e] = (_Float16)b1[e];
  }
  return r;
}

// ---------------------------------------------------------------------------
// Kernel 1: LDS-tiled transpose + f32->f16 convert of the four 512x512 weight
// matrices. Coalesced reads AND writes (32x32 tiles, padded LDS column).
// Output matrix m at ws + m*262144, layout Wt[n][k] = W[k][n].
// ---------------------------------------------------------------------------
__global__ __launch_bounds__(256) void wprep_kernel(const float* __restrict__ Wq,
                                                    const float* __restrict__ Wk,
                                                    const float* __restrict__ Wv,
                                                    const float* __restrict__ Wo,
                                                    _Float16* __restrict__ out) {
  __shared__ float tile[32][33];
  const int bid = blockIdx.x;            // 4 matrices * 256 tiles
  const int m  = bid >> 8;
  const int t  = bid & 255;
  const int k0 = (t & 15) * 32;
  const int n0 = (t >> 4) * 32;
  const float* W = (m == 0) ? Wq : (m == 1) ? Wk : (m == 2) ? Wv : Wo;
  const int tx = threadIdx.x & 31;
  const int ty = threadIdx.x >> 5;       // 0..7
#pragma unroll
  for (int j = 0; j < 4; ++j)
    tile[ty + 8 * j][tx] = W[(size_t)(k0 + ty + 8 * j) * D_MODEL + n0 + tx];
  __syncthreads();
#pragma unroll
  for (int j = 0; j < 4; ++j)
    out[(size_t)m * 262144 + (size_t)(n0 + ty + 8 * j) * D_MODEL + k0 + tx] =
        (_Float16)tile[tx][ty + 8 * j];
}

// ---------------------------------------------------------------------------
// Kernel 2: projection GEMM  Y = enc(f32)[8192,512] @ W[512,512] -> f16.
// mode 0/1: store [b,h,s,dh] row-major (Q, K).  mode 2: store [b,h,dh,s] (V^T).
// One wave computes a 16x64 tile, software-pipelined over the K loop.
// ---------------------------------------------------------------------------
__global__ __launch_bounds__(128) void proj_gemm_kernel(const float* __restrict__ X,
                                                        const _Float16* __restrict__ Wt,
                                                        _Float16* __restrict__ Out,
                                                        int mode) {
  const int wave = blockIdx.x * (blockDim.x >> 5) + (threadIdx.x >> 5);
  const int lane = threadIdx.x & 31;
  const int lm = lane & 15;
  const int h  = lane >> 4;
  const int rowTile = wave >> 3;   // 512 row tiles
  const int colGrp  = wave & 7;    // 8 col groups of 64
  const int i0 = rowTile * 16;
  const int n0 = colGrp * 64;

  v8f acc0 = {}, acc1 = {}, acc2 = {}, acc3 = {};
  const float* xrow = X + (size_t)(i0 + lm) * D_MODEL;
  const _Float16* w0 = Wt + (size_t)(n0 +  0 + lm) * D_MODEL + 8 * h;
  const _Float16* w1 = Wt + (size_t)(n0 + 16 + lm) * D_MODEL + 8 * h;
  const _Float16* w2 = Wt + (size_t)(n0 + 32 + lm) * D_MODEL + 8 * h;
  const _Float16* w3 = Wt + (size_t)(n0 + 48 + lm) * D_MODEL + 8 * h;

  // software pipeline: current fragments in aC/bC*, prefetch next into *N
  v16h aC  = load_a16_f(xrow, 0, h);
  v16h b0C = load_b16(w0);
  v16h b1C = load_b16(w1);
  v16h b2C = load_b16(w2);
  v16h b3C = load_b16(w3);

#pragma unroll 2
  for (int kb = 0; kb < D_MODEL; kb += 32) {
    const int kn = (kb + 32 < D_MODEL) ? (kb + 32) : kb;   // clamped prefetch
    v16h aN  = load_a16_f(xrow, kn, h);
    v16h b0N = load_b16(w0 + kn);
    v16h b1N = load_b16(w1 + kn);
    v16h b2N = load_b16(w2 + kn);
    v16h b3N = load_b16(w3 + kn);

    acc0 = wmma_f16(aC, b0C, acc0);
    acc1 = wmma_f16(aC, b1C, acc1);
    acc2 = wmma_f16(aC, b2C, acc2);
    acc3 = wmma_f16(aC, b3C, acc3);

    aC = aN; b0C = b0N; b1C = b1N; b2C = b2N; b3C = b3N;
  }

#pragma unroll
  for (int r = 0; r < 8; ++r) {
    const int i = i0 + r + 8 * h;          // flat row = b*S + s
    const int b = i >> 12;
    const int s = i & (S_LEN - 1);
#pragma unroll
    for (int j = 0; j < 4; ++j) {
      const int n  = n0 + j * 16 + lm;     // output feature 0..511
      const int hd = n >> 6;
      const int d  = n & 63;
      float v = (j == 0) ? acc0[r] : (j == 1) ? acc1[r] : (j == 2) ? acc2[r] : acc3[r];
      if (mode == 2)
        Out[(((size_t)(b * N_HEADS + hd) * D_HEAD + d) * S_LEN) + s] = (_Float16)v;   // V^T
      else
        Out[(((size_t)(b * N_HEADS + hd) * S_LEN + s) * D_HEAD) + d] = (_Float16)v;   // Q/K
    }
  }
}

// ---------------------------------------------------------------------------
// Kernel 3: causal flash attention, one wave per (b,h,16-query tile).
// Scores computed transposed (S^T = K * Q^T) so the f32 accumulator layout
// is exactly the f16 A-fragment pattern for the P*V WMMA (register-only).
// V fragments are loaded before the softmax math; next chunk's K fragments
// are prefetched before the softmax so VMEM latency hides under VALU work.
// ---------------------------------------------------------------------------
__global__ __launch_bounds__(128) void attn_kernel(const _Float16* __restrict__ Qh,
                                                   const _Float16* __restrict__ Kh,
                                                   const _Float16* __restrict__ Vt,
                                                   _Float16* __restrict__ Ct) {
  const int wave = blockIdx.x * (blockDim.x >> 5) + (threadIdx.x >> 5);
  const int lane = threadIdx.x & 31;
  const int lm = lane & 15;
  const int h  = lane >> 4;
  const int qt = wave & 255;       // 256 q-tiles per (b,h)
  const int bh = wave >> 8;        // 0..15
  const int q0 = qt * 16;
  const int bb = bh >> 3;
  const int hd = bh & 7;

  const _Float16* Qp = Qh + (size_t)bh * S_LEN * D_HEAD;
  const _Float16* Kp = Kh + (size_t)bh * S_LEN * D_HEAD;
  const _Float16* Vp = Vt + (size_t)bh * D_HEAD * S_LEN;

  // Q^T B-fragments for both 32-wide dh halves (lane column = query lm)
  v16h bq0, bq1;
  {
    const _Float16* qrow = Qp + (size_t)(q0 + lm) * D_HEAD;
    bq0 = load_b16(qrow + 0  + 8 * h);
    bq1 = load_b16(qrow + 32 + 8 * h);
  }

  v8f o0 = {}, o1 = {}, o2 = {}, o3 = {};        // out: 16q x 64d (4 tiles)
  float m = -__builtin_inff();
  float l = 0.0f;
  const float scale = 0.125f;                     // 1/sqrt(64)
  const float L2E = 1.44269504088896f;
  const int qg = q0 + lm;                         // query this lane's stats track
  const int kend = q0 + 16;

  // preload K fragments for chunk k0=0:
  // [0]=keys(0..15) d 0..31, [1]=keys(0..15) d 32..63, [2],[3]=keys(16..31)
  v16h ak0, ak1, ak2, ak3;
  {
    const _Float16* kr0 = Kp + (size_t)lm * D_HEAD;
    const _Float16* kr1 = Kp + (size_t)(16 + lm) * D_HEAD;
    ak0 = load_a16_h(kr0, 0, h);  ak1 = load_a16_h(kr0, 32, h);
    ak2 = load_a16_h(kr1, 0, h);  ak3 = load_a16_h(kr1, 32, h);
  }

#pragma unroll 2
  for (int k0 = 0; k0 < kend; k0 += 32) {
    // --- V fragments issued first so their latency hides under softmax VALU
    const _Float16* vbase = Vp + (size_t)lm * S_LEN + k0 + 8 * h;
    v16h bv0 = load_b16(vbase + (size_t)0 * 16 * S_LEN);
    v16h bv1 = load_b16(vbase + (size_t)1 * 16 * S_LEN);
    v16h bv2 = load_b16(vbase + (size_t)2 * 16 * S_LEN);
    v16h bv3 = load_b16(vbase + (size_t)3 * 16 * S_LEN);

    // --- S^T tiles: t0 keys [k0,k0+16), t1 keys [k0+16,k0+32), cols = queries
    v8f t0 = {}, t1 = {};
    t0 = wmma_f16(ak0, bq0, t0);
    t0 = wmma_f16(ak1, bq1, t0);
    t1 = wmma_f16(ak2, bq0, t1);
    t1 = wmma_f16(ak3, bq1, t1);

    // --- prefetch next chunk's K fragments (clamped; discarded on last iter)
    const int k0n = (k0 + 32 < kend) ? (k0 + 32) : k0;
    const _Float16* kr0 = Kp + (size_t)(k0n + lm) * D_HEAD;
    const _Float16* kr1 = Kp + (size_t)(k0n + 16 + lm) * D_HEAD;
    v16h an0 = load_a16_h(kr0, 0, h);
    v16h an1 = load_a16_h(kr0, 32, h);
    v16h an2 = load_a16_h(kr1, 0, h);
    v16h an3 = load_a16_h(kr1, 32, h);

    // --- scale + causal mask + chunk max (lane holds 16 scores of query qg)
    float cmax = -3.0e38f;
#pragma unroll
    for (int r = 0; r < 8; ++r) {
      const int ka = k0 + 8 * h + r;
      const int kb = k0 + 16 + 8 * h + r;
      float sa = t0[r] * scale; if (ka > qg) sa = -3.0e38f;
      float sb = t1[r] * scale; if (kb > qg) sb = -3.0e38f;
      t0[r] = sa; t1[r] = sb;
      cmax = fmaxf(cmax, fmaxf(sa, sb));
    }
    cmax = fmaxf(cmax, __shfl_xor(cmax, 16, 32));
    const float mnew = fmaxf(m, cmax);
    const float alpha = exp2f((m - mnew) * L2E);

    // --- P = exp(S - m); accumulator layout == A-fragment layout
    float lsum = 0.0f;
    v16h ap;
#pragma unroll
    for (int r = 0; r < 8; ++r) {
      float p0 = exp2f((t0[r] - mnew) * L2E);
      float p1 = exp2f((t1[r] - mnew) * L2E);
      lsum += p0 + p1;
      ap[r]     = (_Float16)p0;
      ap[8 + r] = (_Float16)p1;
    }
    lsum += __shfl_xor(lsum, 16, 32);
    l = l * alpha + lsum;
    m = mnew;

    // --- rescale running output (row r <-> query r+8h; stats live in lane q)
    float alr[8];
#pragma unroll
    for (int r = 0; r < 8; ++r) alr[r] = __shfl(alpha, r + 8 * h, 32);
#pragma unroll
    for (int r = 0; r < 8; ++r) {
      o0[r] *= alr[r]; o1[r] *= alr[r]; o2[r] *= alr[r]; o3[r] *= alr[r];
    }

    // --- O += P * V
    o0 = wmma_f16(ap, bv0, o0);
    o1 = wmma_f16(ap, bv1, o1);
    o2 = wmma_f16(ap, bv2, o2);
    o3 = wmma_f16(ap, bv3, o3);

    ak0 = an0; ak1 = an1; ak2 = an2; ak3 = an3;
  }

  // --- finalize: divide by l and store in concat layout [b, s, h*64+d] (f16)
  float rinv[8];
#pragma unroll
  for (int r = 0; r < 8; ++r) {
    float lr = __shfl(l, r + 8 * h, 32);
    rinv[r] = 1.0f / lr;
  }
#pragma unroll
  for (int r = 0; r < 8; ++r) {
    const int s = q0 + r + 8 * h;
    _Float16* crow = Ct + ((size_t)(bb * S_LEN + s) * D_MODEL) + hd * D_HEAD + lm;
    crow[0]  = (_Float16)(o0[r] * rinv[r]);
    crow[16] = (_Float16)(o1[r] * rinv[r]);
    crow[32] = (_Float16)(o2[r] * rinv[r]);
    crow[48] = (_Float16)(o3[r] * rinv[r]);
  }
}

// ---------------------------------------------------------------------------
// Kernel 4: output GEMM  out(f32) = Ct(f16)[8192,512] @ W_O, via Wt_O[n,k].
// Software-pipelined like the projection GEMM.
// ---------------------------------------------------------------------------
__global__ __launch_bounds__(128) void out_gemm_kernel(const _Float16* __restrict__ A,
                                                       const _Float16* __restrict__ Wt,
                                                       float* __restrict__ Out) {
  const int wave = blockIdx.x * (blockDim.x >> 5) + (threadIdx.x >> 5);
  const int lane = threadIdx.x & 31;
  const int lm = lane & 15;
  const int h  = lane >> 4;
  const int rowTile = wave >> 3;
  const int colGrp  = wave & 7;
  const int i0 = rowTile * 16;
  const int n0 = colGrp * 64;

  v8f acc0 = {}, acc1 = {}, acc2 = {}, acc3 = {};
  const _Float16* arow = A + (size_t)(i0 + lm) * D_MODEL;
  const _Float16* w0 = Wt + (size_t)(n0 +  0 + lm) * D_MODEL + 8 * h;
  const _Float16* w1 = Wt + (size_t)(n0 + 16 + lm) * D_MODEL + 8 * h;
  const _Float16* w2 = Wt + (size_t)(n0 + 32 + lm) * D_MODEL + 8 * h;
  const _Float16* w3 = Wt + (size_t)(n0 + 48 + lm) * D_MODEL + 8 * h;

  v16h aC  = load_a16_h(arow, 0, h);
  v16h b0C = load_b16(w0);
  v16h b1C = load_b16(w1);
  v16h b2C = load_b16(w2);
  v16h b3C = load_b16(w3);

#pragma unroll 2
  for (int kb = 0; kb < D_MODEL; kb += 32) {
    const int kn = (kb + 32 < D_MODEL) ? (kb + 32) : kb;
    v16h aN  = load_a16_h(arow, kn, h);
    v16h b0N = load_b16(w0 + kn);
    v16h b1N = load_b16(w1 + kn);
    v16h b2N = load_b16(w2 + kn);
    v16h b3N = load_b16(w3 + kn);

    acc0 = wmma_f16(aC, b0C, acc0);
    acc1 = wmma_f16(aC, b1C, acc1);
    acc2 = wmma_f16(aC, b2C, acc2);
    acc3 = wmma_f16(aC, b3C, acc3);

    aC = aN; b0C = b0N; b1C = b1N; b2C = b2N; b3C = b3N;
  }

#pragma unroll
  for (int r = 0; r < 8; ++r) {
    const int i = i0 + r + 8 * h;
    float* orow = Out + (size_t)i * D_MODEL + n0 + lm;
    orow[0]  = acc0[r];
    orow[16] = acc1[r];
    orow[32] = acc2[r];
    orow[48] = acc3[r];
  }
}

// ---------------------------------------------------------------------------
// Host-side launch
// ---------------------------------------------------------------------------
extern "C" void kernel_launch(void* const* d_in, const int* in_sizes, int n_in,
                              void* d_out, int out_size, void* d_ws, size_t ws_size,
                              hipStream_t stream) {
  (void)in_sizes; (void)n_in; (void)out_size; (void)ws_size;
  const float* enc = (const float*)d_in[0];
  // d_in[1] = causal_mask (int32) — causality is applied analytically, unused.
  const float* Wq = (const float*)d_in[2];
  const float* Wk = (const float*)d_in[3];
  const float* Wv = (const float*)d_in[4];
  const float* Wo = (const float*)d_in[5];
  float* out = (float*)d_out;

  // Workspace layout (all _Float16):
  //   [0, 4*262144)          : WtQ | WtK | WtV | WtO  (transposed f16 weights)
  //   Qh : [b,h,s,dh]   4,194,304 halves
  //   Kh : [b,h,s,dh]   4,194,304 halves
  //   Vt : [b,h,dh,s]   4,194,304 halves
  //   Ct : [b,s,D]      4,194,304 halves
  // total = 17,825,792 halves = ~34 MB
  _Float16* ws  = (_Float16*)d_ws;
  _Float16* WtQ = ws;
  _Float16* WtK = ws + 262144;
  _Float16* WtV = ws + 2 * 262144;
  _Float16* WtO = ws + 3 * 262144;
  _Float16* Qh  = ws + 4 * 262144;
  _Float16* Kh  = Qh + 4194304;
  _Float16* Vt  = Kh + 4194304;
  _Float16* Ct  = Vt + 4194304;

  // 1) weight transpose+convert: 4 matrices * 256 (32x32) tiles
  wprep_kernel<<<1024, 256, 0, stream>>>(Wq, Wk, Wv, Wo, ws);

  // 2) Q/K/V projections: 512 row-tiles * 8 col-groups = 4096 waves, 4/block
  proj_gemm_kernel<<<1024, 128, 0, stream>>>(enc, WtQ, Qh, 0);
  proj_gemm_kernel<<<1024, 128, 0, stream>>>(enc, WtK, Kh, 1);
  proj_gemm_kernel<<<1024, 128, 0, stream>>>(enc, WtV, Vt, 2);

  // 3) causal flash attention: 2*8*256 = 4096 waves, 4/block
  attn_kernel<<<1024, 128, 0, stream>>>(Qh, Kh, Vt, Ct);

  // 4) output projection
  out_gemm_kernel<<<1024, 128, 0, stream>>>(Ct, WtO, out);
}
